// SimpleSelfAttention_56435870269801
// MI455X (gfx1250) — matrix operand order repaired and checked by reading, not compiled, not run on hardware
//
#include <hip/hip_runtime.h>
#include <hip/hip_bf16.h>
#include <math.h>

// ---------------------------------------------------------------------------
// Problem constants (from the reference): x:[4,2048,1024] fp32, E=1024
// ---------------------------------------------------------------------------
#define BATCH 4
#define SEQ   2048
#define EMB   1024
#define MTOT  (BATCH * SEQ)        // 8192 tokens total

typedef __attribute__((ext_vector_type(16))) __bf16 v16bf;
typedef __attribute__((ext_vector_type(8)))  float  v8f;

// Workspace layout (bytes). Total = 134 MiB.
constexpr size_t kMiB   = 1024ull * 1024ull;
constexpr size_t OFF_XB = 0;            // x  in bf16:  8192*1024*2 = 16 MiB
constexpr size_t OFF_WQ = 16 * kMiB;    // Wq in bf16:  2 MiB
constexpr size_t OFF_WK = 18 * kMiB;    // Wk in bf16:  2 MiB
constexpr size_t OFF_WV = 20 * kMiB;    // Wv in bf16:  2 MiB
constexpr size_t OFF_Q  = 22 * kMiB;    // Q  bf16 [B][S][E]: 16 MiB
constexpr size_t OFF_K  = 38 * kMiB;    // K  bf16 [B][S][E]: 16 MiB
constexpr size_t OFF_VT = 54 * kMiB;    // V^T bf16 [B][E][S]: 16 MiB
constexpr size_t OFF_SC = 70 * kMiB;    // scores fp32 [B][S][S]: 64 MiB (P bf16 in place)

// ---------------------------------------------------------------------------
// WMMA fragment loaders (contiguous k order chosen here, same for A and B)
// A 16x32 bf16: lane holds row M=lane&15; e0..15 <- K=k0+16h+0..15 (contiguous)
// B 32x16 bf16: lane holds col N=lane&15; e0..15 <- K=k0+16h+0..15 (contiguous)
// ---------------------------------------------------------------------------
union BF16Frag { uint4 u[2]; v16bf v; };

__device__ __forceinline__ v16bf load_frag_a(const __bf16* __restrict__ base,
                                             size_t row, int k0, size_t pitch, int half) {
  const char* p = (const char*)(base + row * pitch + (size_t)k0 + half * 16);
  BF16Frag f;
  f.u[0] = *(const uint4*)(p);        // K = k0 + 16h + 0..7
  f.u[1] = *(const uint4*)(p + 16);   // K = k0 + 16h + 8..15
  return f.v;
}

__device__ __forceinline__ v16bf load_frag_b(const __bf16* __restrict__ base,
                                             size_t row, int k0, size_t pitch, int half) {
  const char* p = (const char*)(base + row * pitch + (size_t)k0 + half * 16);
  BF16Frag f;
  f.u[0] = *(const uint4*)(p);        // K = k0 + 16h + 0..7
  f.u[1] = *(const uint4*)(p + 16);   // K = k0 + 16h + 8..15
  return f.v;
}

__device__ __forceinline__ v8f wmma_bf16(v16bf a, v16bf b, v8f c) {
  return __builtin_amdgcn_wmma_f32_16x16x32_bf16(false, a, false, b, (short)0, c,
                                                 false, false);
}

// ---------------------------------------------------------------------------
// Kernel 1: fp32 -> bf16 conversion (x and the three weight matrices)
// ---------------------------------------------------------------------------
__global__ void cvt_f32_to_bf16(const float* __restrict__ src,
                                __bf16* __restrict__ dst, int n4) {
  int i = blockIdx.x * blockDim.x + threadIdx.x;
  if (i < n4) {
    float4 f = ((const float4*)src)[i];
    union { __bf16 h[4]; uint2 u; } o;
    o.h[0] = (__bf16)f.x; o.h[1] = (__bf16)f.y;
    o.h[2] = (__bf16)f.z; o.h[3] = (__bf16)f.w;
    ((uint2*)dst)[i] = o.u;
  }
}

// ---------------------------------------------------------------------------
// Kernel 2: QKV projection.  y[m, f] = sum_e x[m,e] * W[f,e] + bias[f]
// blockIdx.z selects Q / K / V.  One wave per 32x32 output block (2x2 WMMA
// tiles): each A fragment feeds 2 B tiles and vice versa -> 2 loads/WMMA.
// V is stored transposed (Vt[b][f][s]) so the PV GEMM gets contiguous B rows.
// ---------------------------------------------------------------------------
__global__ void qkv_gemm(const __bf16* __restrict__ xb,
                         const __bf16* __restrict__ wq, const __bf16* __restrict__ wk,
                         const __bf16* __restrict__ wv,
                         const float* __restrict__ bq, const float* __restrict__ bk,
                         const float* __restrict__ bv,
                         __bf16* __restrict__ Qb, __bf16* __restrict__ Kb,
                         __bf16* __restrict__ Vt) {
  const int which = blockIdx.z;
  const __bf16* W   = (which == 0) ? wq : (which == 1) ? wk : wv;
  const float* bias = (which == 0) ? bq : (which == 1) ? bk : bv;

  const int wave = (blockIdx.x * blockDim.x + threadIdx.x) >> 5;
  const int lane = threadIdx.x & 31;
  const int half = lane >> 4;
  const int blkN = EMB / 32;                   // 32
  const int bn = wave % blkN;
  const int bm = wave / blkN;                  // < MTOT/32 = 256
  const int m0 = bm * 32, n0 = bn * 32;

  const size_t rA0 = m0 +      (lane & 15);
  const size_t rA1 = m0 + 16 + (lane & 15);
  const size_t rB0 = n0 +      (lane & 15);
  const size_t rB1 = n0 + 16 + (lane & 15);

  v8f acc[2][2] = {};
  #pragma unroll 2
  for (int k0 = 0; k0 < EMB; k0 += 32) {
    v16bf a0 = load_frag_a(xb, rA0, k0, EMB, half);
    v16bf a1 = load_frag_a(xb, rA1, k0, EMB, half);
    v16bf b0 = load_frag_b(W,  rB0, k0, EMB, half);
    v16bf b1 = load_frag_b(W,  rB1, k0, EMB, half);
    acc[0][0] = wmma_bf16(a0, b0, acc[0][0]);
    acc[0][1] = wmma_bf16(a0, b1, acc[0][1]);
    acc[1][0] = wmma_bf16(a1, b0, acc[1][0]);
    acc[1][1] = wmma_bf16(a1, b1, acc[1][1]);
  }

  #pragma unroll
  for (int mi = 0; mi < 2; ++mi) {
    #pragma unroll
    for (int ni = 0; ni < 2; ++ni) {
      const int   n  = n0 + ni * 16 + (lane & 15);
      const float bf = bias[n];
      if (which < 2) {
        __bf16* dst = (which == 0) ? Qb : Kb;  // [B*S][E] row-major
        #pragma unroll
        for (int r = 0; r < 8; ++r) {
          int m = m0 + mi * 16 + half * 8 + r;
          dst[(size_t)m * EMB + n] = (__bf16)(acc[mi][ni][r] + bf);
        }
      } else {
        // Vt[b][n][s_local]; per-lane 8 consecutive s -> one b128 store
        const int bb = m0 / SEQ;               // 32-blocks never straddle batches
        const int s0 = m0 + mi * 16 + half * 8 - bb * SEQ;
        union { __bf16 h[8]; uint4 u; } t;
        #pragma unroll
        for (int r = 0; r < 8; ++r) t.h[r] = (__bf16)(acc[mi][ni][r] + bf);
        *(uint4*)(Vt + ((size_t)bb * EMB + n) * SEQ + s0) = t.u;
      }
    }
  }
}

// ---------------------------------------------------------------------------
// Kernel 3: scores S[b,q,k] = (Q[b,q,:] . K[b,k,:]) / sqrt(E)
// One wave per 32x32 score block; B-fragment column k = row k of K (contiguous).
// ---------------------------------------------------------------------------
__global__ void scores_gemm(const __bf16* __restrict__ Qb,
                            const __bf16* __restrict__ Kb,
                            float* __restrict__ Sc) {
  const int wave = (blockIdx.x * blockDim.x + threadIdx.x) >> 5;
  const int lane = threadIdx.x & 31;
  const int half = lane >> 4;
  const int blk  = SEQ / 32;                   // 64
  const int perB = blk * blk;                  // 4096 blocks per batch
  const int b  = wave / perB;
  const int t  = wave % perB;
  const int bk = t % blk;
  const int bq = t / blk;
  const int q0 = bq * 32, n0 = bk * 32;

  const __bf16* Qr = Qb + (size_t)b * SEQ * EMB;
  const __bf16* Kr = Kb + (size_t)b * SEQ * EMB;
  const size_t rA0 = q0 +      (lane & 15);
  const size_t rA1 = q0 + 16 + (lane & 15);
  const size_t rB0 = n0 +      (lane & 15);
  const size_t rB1 = n0 + 16 + (lane & 15);

  v8f acc[2][2] = {};
  #pragma unroll 2
  for (int k0 = 0; k0 < EMB; k0 += 32) {
    v16bf a0 = load_frag_a(Qr, rA0, k0, EMB, half);
    v16bf a1 = load_frag_a(Qr, rA1, k0, EMB, half);
    v16bf b0 = load_frag_b(Kr, rB0, k0, EMB, half);
    v16bf b1 = load_frag_b(Kr, rB1, k0, EMB, half);
    acc[0][0] = wmma_bf16(a0, b0, acc[0][0]);
    acc[0][1] = wmma_bf16(a0, b1, acc[0][1]);
    acc[1][0] = wmma_bf16(a1, b0, acc[1][0]);
    acc[1][1] = wmma_bf16(a1, b1, acc[1][1]);
  }

  const float scale = 0.03125f;                // 1/sqrt(1024)
  float* S = Sc + (size_t)b * SEQ * SEQ;
  #pragma unroll
  for (int mi = 0; mi < 2; ++mi) {
    #pragma unroll
    for (int ni = 0; ni < 2; ++ni) {
      const int n = n0 + ni * 16 + (lane & 15);
      #pragma unroll
      for (int r = 0; r < 8; ++r) {
        int m = q0 + mi * 16 + half * 8 + r;
        S[(size_t)m * SEQ + n] = acc[mi][ni][r] * scale;
      }
    }
  }
}

// ---------------------------------------------------------------------------
// Kernel 4: softmax over each score row (length 2048); writes bf16 P in place
// (bf16 row occupies the first half of the fp32 row's 8 KB).
// ---------------------------------------------------------------------------
__global__ void softmax_rows(float* __restrict__ Sc) {
  __shared__ float sred[8];
  const int row = blockIdx.x;                  // B*S rows
  float* rp = Sc + (size_t)row * SEQ;
  const int t = threadIdx.x;                   // 256 threads, 8 elems each

  float v[8];
  float m = -3.0e38f;
  #pragma unroll
  for (int i = 0; i < 8; ++i) { v[i] = rp[t + i * 256]; m = fmaxf(m, v[i]); }
  #pragma unroll
  for (int o = 16; o >= 1; o >>= 1) m = fmaxf(m, __shfl_xor(m, o, 32));
  if ((t & 31) == 0) sred[t >> 5] = m;
  __syncthreads();
  m = sred[0];
  #pragma unroll
  for (int i = 1; i < 8; ++i) m = fmaxf(m, sred[i]);
  __syncthreads();

  float s = 0.0f;
  #pragma unroll
  for (int i = 0; i < 8; ++i) { v[i] = __expf(v[i] - m); s += v[i]; }
  #pragma unroll
  for (int o = 16; o >= 1; o >>= 1) s += __shfl_xor(s, o, 32);
  if ((t & 31) == 0) sred[t >> 5] = s;
  __syncthreads();
  s = 0.0f;
  #pragma unroll
  for (int i = 0; i < 8; ++i) s += sred[i];
  const float inv = 1.0f / s;

  __bf16* op = (__bf16*)rp;                    // in-place: all reads done (barriers above)
  #pragma unroll
  for (int i = 0; i < 8; ++i) op[t + i * 256] = (__bf16)(v[i] * inv);
}

// ---------------------------------------------------------------------------
// Kernel 5: O[b,q,d] = sum_k P[b,q,k] * V[b,k,d]  (B from Vt rows, contiguous)
// P rows are bf16 with pitch SEQ*2 bf16 elements (= the fp32 row pitch).
// One wave per 32x32 output block.
// ---------------------------------------------------------------------------
__global__ void pv_gemm(const float* __restrict__ Sc,
                        const __bf16* __restrict__ Vt,
                        float* __restrict__ out) {
  const int wave = (blockIdx.x * blockDim.x + threadIdx.x) >> 5;
  const int lane = threadIdx.x & 31;
  const int half = lane >> 4;
  const int blkD = EMB / 32;                   // 32
  const int perB = (SEQ / 32) * blkD;          // 2048
  const int b  = wave / perB;
  const int t  = wave % perB;
  const int bd = t % blkD;
  const int bq = t / blkD;
  const int q0 = bq * 32, d0 = bd * 32;

  const __bf16* P = (const __bf16*)(Sc + (size_t)b * SEQ * SEQ);
  const size_t pitchP = (size_t)SEQ * 2;       // bf16 elements between row starts
  const __bf16* V = Vt + (size_t)b * EMB * SEQ;
  const size_t rA0 = q0 +      (lane & 15);
  const size_t rA1 = q0 + 16 + (lane & 15);
  const size_t rB0 = d0 +      (lane & 15);
  const size_t rB1 = d0 + 16 + (lane & 15);

  v8f acc[2][2] = {};
  #pragma unroll 2
  for (int k0 = 0; k0 < SEQ; k0 += 32) {
    v16bf a0 = load_frag_a(P, rA0, k0, pitchP, half);
    v16bf a1 = load_frag_a(P, rA1, k0, pitchP, half);
    v16bf b0 = load_frag_b(V, rB0, k0, SEQ, half);
    v16bf b1 = load_frag_b(V, rB1, k0, SEQ, half);
    acc[0][0] = wmma_bf16(a0, b0, acc[0][0]);
    acc[0][1] = wmma_bf16(a0, b1, acc[0][1]);
    acc[1][0] = wmma_bf16(a1, b0, acc[1][0]);
    acc[1][1] = wmma_bf16(a1, b1, acc[1][1]);
  }

  float* O = out + (size_t)b * SEQ * EMB;
  #pragma unroll
  for (int mi = 0; mi < 2; ++mi) {
    #pragma unroll
    for (int ni = 0; ni < 2; ++ni) {
      const int n = d0 + ni * 16 + (lane & 15);
      #pragma unroll
      for (int r = 0; r < 8; ++r)
        O[(size_t)(q0 + mi * 16 + half * 8 + r) * EMB + n] = acc[mi][ni][r];
    }
  }
}

// ---------------------------------------------------------------------------
// Launch
// ---------------------------------------------------------------------------
extern "C" void kernel_launch(void* const* d_in, const int* in_sizes, int n_in,
                              void* d_out, int out_size, void* d_ws, size_t ws_size,
                              hipStream_t stream) {
  (void)in_sizes; (void)n_in; (void)out_size; (void)ws_size;
  const float* x  = (const float*)d_in[0];
  const float* Wq = (const float*)d_in[1];
  const float* bq = (const float*)d_in[2];
  const float* Wk = (const float*)d_in[3];
  const float* bk = (const float*)d_in[4];
  const float* Wv = (const float*)d_in[5];
  const float* bv = (const float*)d_in[6];
  float* out = (float*)d_out;

  char* ws = (char*)d_ws;
  __bf16* xb  = (__bf16*)(ws + OFF_XB);
  __bf16* wqb = (__bf16*)(ws + OFF_WQ);
  __bf16* wkb = (__bf16*)(ws + OFF_WK);
  __bf16* wvb = (__bf16*)(ws + OFF_WV);
  __bf16* Qb  = (__bf16*)(ws + OFF_Q);
  __bf16* Kb  = (__bf16*)(ws + OFF_K);
  __bf16* Vt  = (__bf16*)(ws + OFF_VT);
  float*  Sc  = (float*)(ws + OFF_SC);

  // 1) fp32 -> bf16
  cvt_f32_to_bf16<<<(MTOT * EMB / 4) / 256, 256, 0, stream>>>(x,  xb,  MTOT * EMB / 4);
  cvt_f32_to_bf16<<<(EMB * EMB / 4) / 256,  256, 0, stream>>>(Wq, wqb, EMB * EMB / 4);
  cvt_f32_to_bf16<<<(EMB * EMB / 4) / 256,  256, 0, stream>>>(Wk, wkb, EMB * EMB / 4);
  cvt_f32_to_bf16<<<(EMB * EMB / 4) / 256,  256, 0, stream>>>(Wv, wvb, EMB * EMB / 4);

  // 2) QKV projection: (8192/32)*(1024/32) = 8192 waves per output, 8 waves/block
  qkv_gemm<<<dim3((MTOT / 32) * (EMB / 32) / 8, 1, 3), 256, 0, stream>>>(
      xb, wqb, wkb, wvb, bq, bk, bv, Qb, Kb, Vt);

  // 3) scores: 4 * 64 * 64 = 16384 waves
  scores_gemm<<<BATCH * (SEQ / 32) * (SEQ / 32) / 8, 256, 0, stream>>>(Qb, Kb, Sc);

  // 4) softmax per row (8192 rows)
  softmax_rows<<<BATCH * SEQ, 256, 0, stream>>>(Sc);

  // 5) P @ V: 4 * 64 * 32 = 8192 waves
  pv_gemm<<<BATCH * (SEQ / 32) * (EMB / 32) / 8, 256, 0, stream>>>(Sc, Vt, out);
}
